// XdiffPerciever_20744692039776
// MI455X (gfx1250) — matrix operand order repaired and checked by reading, not compiled, not run on hardware
//
#include <hip/hip_runtime.h>
#include <hip/hip_bf16.h>
#include <math.h>

// ---------------- model constants (match reference) ----------------
#define B_    2
#define L_    256
#define N_    2048
#define DIN_  16
#define DM_   128
#define H_    8
#define K_    10
#define F_    21      // 2K+1
#define FP_   32      // F padded for f16 WMMA K-step
#define DFF_  512
#define PI_F  3.14159265358979323846f

typedef __attribute__((ext_vector_type(16))) _Float16 v16h;
typedef __attribute__((ext_vector_type(8)))  _Float16 v8h;
typedef __attribute__((ext_vector_type(8)))  float    v8f;

// Branchless exact-GELU: erf via Abramowitz-Stegun 7.1.26 (|err| <= 2.5e-5),
// straight-line code: rcp + 5 fma + hw exp + bfi, no exec-mask divergence.
__device__ __forceinline__ float gelu_fast(float x) {
  float z  = x * 0.70710678118654752f;     // x / sqrt(2)
  float az = fabsf(z);
  float t  = __builtin_amdgcn_rcpf(fmaf(0.3275911f, az, 1.0f));
  float y  = fmaf(t, 1.061405429f, -1.453152027f);
  y = fmaf(t, y, 1.421413741f);
  y = fmaf(t, y, -0.284496736f);
  y = fmaf(t, y, 0.254829592f);
  y = y * t;
  float e  = __expf(-az * az);             // v_exp_f32
  float er = fmaf(-y, e, 1.0f);            // erf(|z|)
  er = copysignf(er, z);
  return 0.5f * x * (1.0f + er);
}

// ---- CDNA5 WMMA fragment loaders ----
// A layout: 16x32 (MxK) f16. lanes 0-15: m=lane,    K = k0+{0..7, 16..23}
//                            lanes 16-31: m=lane-16, K = k0+{8..15, 24..31}
// B layout: 32x16 (KxN) f16. lanes 0-15: n=lane,    K = k0+{0..7, 16..23}
//                            lanes 16-31: n=lane-16, K = k0+{8..15, 24..31}

// A from f32, full 32-wide K step: 4x b128 + packed converts.
__device__ __forceinline__ v16h load_frag_a_fast(const float* __restrict__ A,
                                                 int lda, int m0, int k0, int lane) {
  int m    = m0 + (lane & 15);
  int base = k0 + ((lane >> 4) << 3);
  const float* p = A + (size_t)m * lda + base;
  float4 r0 = *(const float4*)(p);
  float4 r1 = *(const float4*)(p + 4);
  float4 r2 = *(const float4*)(p + 16);
  float4 r3 = *(const float4*)(p + 20);
  v16h a;
  a[0]  = (_Float16)r0.x; a[1]  = (_Float16)r0.y; a[2]  = (_Float16)r0.z; a[3]  = (_Float16)r0.w;
  a[4]  = (_Float16)r1.x; a[5]  = (_Float16)r1.y; a[6]  = (_Float16)r1.z; a[7]  = (_Float16)r1.w;
  a[8]  = (_Float16)r2.x; a[9]  = (_Float16)r2.y; a[10] = (_Float16)r2.z; a[11] = (_Float16)r2.w;
  a[12] = (_Float16)r3.x; a[13] = (_Float16)r3.y; a[14] = (_Float16)r3.z; a[15] = (_Float16)r3.w;
  return a;
}

// A from f32, K edge (K=16 only): clamped loads + cndmask, no exec branching.
__device__ __forceinline__ v16h load_frag_a_edge(const float* __restrict__ A,
                                                 int lda, int K, int m0, int k0, int lane) {
  int m    = m0 + (lane & 15);
  int base = k0 + ((lane >> 4) << 3);
  const float* row = A + (size_t)m * lda;
  v16h a;
#pragma unroll
  for (int j = 0; j < 16; ++j) {
    int k  = base + (j < 8 ? j : j + 8);
    int kc = k < K ? k : K - 1;
    float v = row[kc];
    a[j] = (_Float16)(k < K ? v : 0.0f);
  }
  return a;
}

// A from pre-packed f16 rows (row stride ldah halves, K pre-padded): 2x b128, no converts.
__device__ __forceinline__ v16h load_frag_a_h(const _Float16* __restrict__ Ah,
                                              int ldah, int m0, int k0, int lane) {
  int m    = m0 + (lane & 15);
  int base = k0 + ((lane >> 4) << 3);
  const _Float16* p = Ah + (size_t)m * ldah + base;
  v8h r0 = *(const v8h*)(p);        // K = base..base+7
  v8h r1 = *(const v8h*)(p + 16);   // K = base+16..base+23
  v16h a;
#pragma unroll
  for (int j = 0; j < 8; ++j) { a[j] = r0[j]; a[j + 8] = r1[j]; }
  return a;
}

// B from pre-transposed/converted weights Wt[Np][Kp] f16 (zero padded): 2x b128.
__device__ __forceinline__ v16h load_frag_b_t(const _Float16* __restrict__ Wt,
                                              int Kp, int k0, int n0, int lane) {
  int n  = n0 + (lane & 15);
  int kb = k0 + ((lane >> 4) << 3);
  const _Float16* p = Wt + (size_t)n * Kp + kb;
  v8h r0 = *(const v8h*)(p);
  v8h r1 = *(const v8h*)(p + 16);
  v16h b;
#pragma unroll
  for (int j = 0; j < 8; ++j) { b[j] = r0[j]; b[j + 8] = r1[j]; }
  return b;
}

// ---- weight transform: W[K,N] f32 row-major -> Wt[Np,Kp] f16, zero padded ----
__global__ __launch_bounds__(256)
void wt_kernel(const float* __restrict__ W, _Float16* __restrict__ Wt,
               int K, int N, int Kp, int Np) {
  int id = blockIdx.x * 256 + threadIdx.x;
  if (id >= Kp * Np) return;
  int k = id % Kp;
  int n = id / Kp;
  float v = (k < K && n < N) ? W[(size_t)k * N + n] : 0.0f;
  Wt[(size_t)n * Kp + k] = (_Float16)v;
}

// ---------------- xdiff: distance + sin/cos, packed f16 rows of 32 ----------------
__global__ __launch_bounds__(256)
void xdiff_kernel(const float* __restrict__ x1, const float* __restrict__ x2,
                  _Float16* __restrict__ out, int n1, int n2) {
  int id = blockIdx.x * 256 + threadIdx.x;
  int tot = B_ * n1 * n2;
  if (id >= tot) return;
  int j = id % n2;
  int i = (id / n2) % n1;
  int b = id / (n1 * n2);
  const float* p1 = x1 + ((size_t)b * n1 + i) * 3;
  const float* p2 = x2 + ((size_t)b * n2 + j) * 3;
  float dx = p1[0] - p2[0], dy = p1[1] - p2[1], dz = p1[2] - p2[2];
  float d = sqrtf(dx * dx + dy * dy + dz * dz);   // SCALE == 1
  float vals[FP_];
  vals[0] = d;
#pragma unroll
  for (int k = 1; k <= K_; ++k) {
    float s, c;
    sincosf(PI_F * (float)k * d, &s, &c);
    vals[2 * k - 1] = s;
    vals[2 * k]     = c;
  }
#pragma unroll
  for (int f = F_; f < FP_; ++f) vals[f] = 0.0f;
  v8h h0, h1, h2, h3;
#pragma unroll
  for (int f = 0; f < 8; ++f) {
    h0[f] = (_Float16)vals[f];
    h1[f] = (_Float16)vals[f + 8];
    h2[f] = (_Float16)vals[f + 16];
    h3[f] = (_Float16)vals[f + 24];
  }
  _Float16* op = out + (size_t)id * FP_;
  *(v8h*)(op)      = h0;
  *(v8h*)(op + 8)  = h1;
  *(v8h*)(op + 16) = h2;
  *(v8h*)(op + 24) = h3;
}

// ---- fused x_token: out[b,l,c] (+)= mean_n gelu(xd[b,l,n,:] @ W + bias) ----
// grid = B*L blocks, 8 waves; wave w owns channels [16w,16w+16).
__global__ __launch_bounds__(256)
void token_kernel(const _Float16* __restrict__ xdh, const _Float16* __restrict__ Wt,
                  const float* __restrict__ bias, float* __restrict__ out,
                  int rows, int beta) {
  int bl   = blockIdx.x;
  int wid  = threadIdx.x >> 5;
  int lane = threadIdx.x & 31;
  int n0   = wid * 16;
  const _Float16* Ab = xdh + (size_t)bl * rows * FP_;
  v16h bf = load_frag_b_t(Wt, FP_, 0, n0, lane);
  int col = n0 + (lane & 15);
  float bb = bias[col];
  float s = 0.0f;
  for (int m0 = 0; m0 < rows; m0 += 16) {   // rows is a multiple of 16
    v16h a = load_frag_a_h(Ab, FP_, m0, 0, lane);
    v8f c = {};
    c = __builtin_amdgcn_wmma_f32_16x16x32_f16(false, a, false, bf,
                                               (short)0, c, false, false);
#pragma unroll
    for (int r = 0; r < 8; ++r) s += gelu_fast(c[r] + bb);
  }
  s += __shfl_xor(s, 16, 32);           // merge the two m-halves (same column)
  if (lane < 16) {
    size_t idx = (size_t)bl * DM_ + col;
    float val = s / (float)rows;
    out[idx] = beta ? (out[idx] + val) : val;
  }
}

// ---------------- generic WMMA GEMM: C = [res +] act(A@Wt + bias) ----------------
// A row-major [M,K] f32; Wt pre-converted [Np,Kp] f16. One wave per 16x16 tile.
// M must be a multiple of 16 (true for all call sites).
__global__ __launch_bounds__(32)
void wmma_gemm(const float* __restrict__ A, const _Float16* __restrict__ Wt,
               const float* __restrict__ bias, const float* __restrict__ res,
               float* __restrict__ C, int M, int N, int K, int Kp, int act) {
  int lane = threadIdx.x;
  int m0 = blockIdx.x * 16;
  int n0 = blockIdx.y * 16;
  bool nfull = (n0 + 16 <= N);          // wave-uniform
  v8f acc = {};
  for (int k0 = 0; k0 < K; k0 += 32) {
    v16h a = (k0 + 32 <= K) ? load_frag_a_fast(A, K, m0, k0, lane)
                            : load_frag_a_edge(A, K, K, m0, k0, lane);
    v16h b = load_frag_b_t(Wt, Kp, k0, n0, lane);
    acc = __builtin_amdgcn_wmma_f32_16x16x32_f16(false, a, false, b,
                                                 (short)0, acc, false, false);
  }
  int col = n0 + (lane & 15);
  int mb  = m0 + ((lane >> 4) << 3);
  float bv = bias ? bias[nfull ? col : (col < N ? col : N - 1)] : 0.0f;
  if (nfull || col < N) {
#pragma unroll
    for (int r = 0; r < 8; ++r) {
      int row = mb + r;
      float v = acc[r] + bv;
      if (act) v = gelu_fast(v);
      size_t idx = (size_t)row * N + col;
      if (res) v += res[idx];
      C[idx] = v;
    }
  }
}

// ---------------- LayerNorm over d=128, one wave per row ----------------
__global__ __launch_bounds__(32)
void ln_kernel(const float* __restrict__ x, const float* __restrict__ g,
               const float* __restrict__ bt, float* __restrict__ y, int rows) {
  int row = blockIdx.x;
  int lane = threadIdx.x;
  const float* xr = x + (size_t)row * DM_ + lane * 4;
  float4 v = *(const float4*)xr;
  float s = v.x + v.y + v.z + v.w;
#pragma unroll
  for (int off = 16; off > 0; off >>= 1) s += __shfl_xor(s, off, 32);
  float mu = s * (1.0f / DM_);
  float dx = v.x - mu, dy = v.y - mu, dz = v.z - mu, dw = v.w - mu;
  float q = dx * dx + dy * dy + dz * dz + dw * dw;
#pragma unroll
  for (int off = 16; off > 0; off >>= 1) q += __shfl_xor(q, off, 32);
  float inv = rsqrtf(q * (1.0f / DM_) + 1e-6f);
  float4 gg = *(const float4*)(g + lane * 4);
  float4 bb = *(const float4*)(bt + lane * 4);
  float4 o;
  o.x = gg.x * dx * inv + bb.x;
  o.y = gg.y * dy * inv + bb.y;
  o.z = gg.z * dz * inv + bb.z;
  o.w = gg.w * dw * inv + bb.w;
  *(float4*)(y + (size_t)row * DM_ + lane * 4) = o;
}

// ---- fused attention: one block per (b,h,query-row); logits = qk/4 + xdot/sqrt(21) ----
__global__ __launch_bounds__(256)
void attn_kernel(const float* __restrict__ qp, const float* __restrict__ kp,
                 const float* __restrict__ vp, const float* __restrict__ xw,
                 const _Float16* __restrict__ xdh, float* __restrict__ o,
                 int Lq, int Nk) {
  __shared__ float qv[16];
  __shared__ float xv[F_];
  __shared__ float logits[N_];
  __shared__ float red[256];
  int id = blockIdx.x;
  int l = id % Lq;
  int h = (id / Lq) % H_;
  int b = id / (Lq * H_);
  int t = threadIdx.x;
  size_t qrow = (size_t)(b * Lq + l);
  if (t < 16)  qv[t] = qp[qrow * DM_ + h * 16 + t];
  if (t < F_)  xv[t] = xw[qrow * (H_ * F_) + h * F_ + t];
  __syncthreads();

  const float qk_scale = 0.25f;                 // 1/sqrt(depth=16)
  const float x_scale  = 0.21821789023599238f;  // 1/sqrt(21)
  float lmax = -3.0e38f;
  for (int n = t; n < Nk; n += 256) {
    const float* kr = kp + ((size_t)(b * Nk + n)) * DM_ + h * 16;
    float4 k0 = *(const float4*)(kr);
    float4 k1 = *(const float4*)(kr + 4);
    float4 k2 = *(const float4*)(kr + 8);
    float4 k3 = *(const float4*)(kr + 12);
    float dq = qv[0] * k0.x + qv[1] * k0.y + qv[2] * k0.z + qv[3] * k0.w
             + qv[4] * k1.x + qv[5] * k1.y + qv[6] * k1.z + qv[7] * k1.w
             + qv[8] * k2.x + qv[9] * k2.y + qv[10] * k2.z + qv[11] * k2.w
             + qv[12] * k3.x + qv[13] * k3.y + qv[14] * k3.z + qv[15] * k3.w;
    const _Float16* xr = xdh + (qrow * Nk + n) * FP_;
    v8h x0 = *(const v8h*)(xr);
    v8h x1 = *(const v8h*)(xr + 8);
    v8h x2 = *(const v8h*)(xr + 16);
    float dxv = 0.0f;
#pragma unroll
    for (int f = 0; f < 8; ++f) dxv += xv[f] * (float)x0[f];
#pragma unroll
    for (int f = 0; f < 8; ++f) dxv += xv[f + 8] * (float)x1[f];
#pragma unroll
    for (int f = 0; f < 5; ++f) dxv += xv[f + 16] * (float)x2[f];
    float lg = dq * qk_scale + dxv * x_scale;
    logits[n] = lg;
    lmax = fmaxf(lmax, lg);
  }
  red[t] = lmax; __syncthreads();
  for (int s = 128; s > 0; s >>= 1) {
    if (t < s) red[t] = fmaxf(red[t], red[t + s]);
    __syncthreads();
  }
  float mx = red[0]; __syncthreads();

  float vacc[16];
#pragma unroll
  for (int d = 0; d < 16; ++d) vacc[d] = 0.0f;
  float lsum = 0.0f;
  for (int n = t; n < Nk; n += 256) {
    float p = __expf(logits[n] - mx);           // hardware v_exp_f32
    lsum += p;
    const float* vr = vp + ((size_t)(b * Nk + n)) * DM_ + h * 16;
    float4 u0 = *(const float4*)(vr);
    float4 u1 = *(const float4*)(vr + 4);
    float4 u2 = *(const float4*)(vr + 8);
    float4 u3 = *(const float4*)(vr + 12);
    vacc[0]  += p * u0.x; vacc[1]  += p * u0.y; vacc[2]  += p * u0.z; vacc[3]  += p * u0.w;
    vacc[4]  += p * u1.x; vacc[5]  += p * u1.y; vacc[6]  += p * u1.z; vacc[7]  += p * u1.w;
    vacc[8]  += p * u2.x; vacc[9]  += p * u2.y; vacc[10] += p * u2.z; vacc[11] += p * u2.w;
    vacc[12] += p * u3.x; vacc[13] += p * u3.y; vacc[14] += p * u3.z; vacc[15] += p * u3.w;
  }
  red[t] = lsum; __syncthreads();
  for (int s = 128; s > 0; s >>= 1) {
    if (t < s) red[t] += red[t + s];
    __syncthreads();
  }
  float denom = red[0]; __syncthreads();

  for (int d = 0; d < 16; ++d) {
    red[t] = vacc[d]; __syncthreads();
    for (int s = 128; s > 0; s >>= 1) {
      if (t < s) red[t] += red[t + s];
      __syncthreads();
    }
    if (t == 0) o[qrow * DM_ + h * 16 + d] = red[0] / denom;
    __syncthreads();
  }
}

// ---------------- host-side orchestration ----------------
extern "C" void kernel_launch(void* const* d_in, const int* in_sizes, int n_in,
                              void* d_out, int out_size, void* d_ws, size_t ws_size,
                              hipStream_t stream) {
  (void)in_sizes; (void)n_in; (void)out_size; (void)ws_size;
  auto P = [&](int i) { return (const float*)d_in[i]; };

  // Leaf order: setup_inputs() dict insertion order, recursive.
  // 0 x_latents, 1 x_inputs, 2 inputs, 3 wxl_w, 4 wxl_b, 5 wxi_w, 6 wxi_b,
  // cross r: base 7+22r: inp_w,inp_b,ln1a_g,ln1a_b,ln1b_g,ln1b_b,
  //   wq_w,wq_b,wk_w,wk_b,wv_w,wv_b,wx_w,wx_b,wo_w,wo_b,ln2_g,ln2_b,
  //   ffn0_w,ffn0_b,ffn1_w,ffn1_b
  // enc[r][l]: base 51+18*(2r+l): ln1_g,ln1_b,wq_w,wq_b,wk_w,wk_b,wv_w,wv_b,
  //   wx_w,wx_b,wo_w,wo_b,ln2_g,ln2_b,ffn0_w,ffn0_b,ffn1_w,ffn1_b
  // 123 ln_f_g, 124 ln_f_b, 125 pre_w, 126 pre_b, 127 final_w, 128 final_b
  const float* x_lat  = P(0);
  const float* x_inp  = P(1);
  const float* inputs = P(2);

  // -------- workspace layout: f32 activations, then f16 region --------
  float* ws   = (float*)d_ws;
  float* lat  = ws;                                   // [B*L,128]
  float* latn = lat  + (size_t)B_ * L_ * DM_;
  float* inp  = latn + (size_t)B_ * L_ * DM_;         // [B*N,128]
  float* inpn = inp  + (size_t)B_ * N_ * DM_;
  float* qp   = inpn + (size_t)B_ * N_ * DM_;         // [B*L,128]
  float* kp   = qp   + (size_t)B_ * L_ * DM_;         // [B*N,128]
  float* vp   = kp   + (size_t)B_ * N_ * DM_;
  float* xw   = vp   + (size_t)B_ * N_ * DM_;         // [B*L,168]
  float* ob   = xw   + (size_t)B_ * L_ * H_ * F_;     // [B*L,128]
  float* h1   = ob   + (size_t)B_ * L_ * DM_;         // [B*L,512]
  _Float16* wsh  = (_Float16*)(h1 + (size_t)B_ * L_ * DFF_);
  _Float16* xdsh = wsh;                               // [B,L,L,32] f16
  _Float16* xdch = xdsh + (size_t)B_ * L_ * L_ * FP_; // [B,L,N,32] f16
  _Float16* wreg = xdch + (size_t)B_ * L_ * N_ * FP_; // converted weights
  size_t hoff = 0;

  // convert W[K,N] f32 -> Wt[Np,Kp] f16 (zero padded); returns device pointer
  auto convW = [&](int leaf, int K, int N) -> const _Float16* {
    int Kp = ((K + 31) / 32) * 32;
    int Np = ((N + 15) / 16) * 16;
    _Float16* dst = wreg + hoff;
    hoff += (size_t)Kp * Np;
    int tot = Kp * Np;
    wt_kernel<<<(tot + 255) / 256, 256, 0, stream>>>(P(leaf), dst, K, N, Kp, Np);
    return dst;
  };

  auto gemm = [&](const float* A, const _Float16* Wt, const float* bias,
                  const float* res, float* C, int M, int N, int K, int act) {
    int Kp = ((K + 31) / 32) * 32;
    dim3 grid((M + 15) / 16, (N + 15) / 16);
    wmma_gemm<<<grid, 32, 0, stream>>>(A, Wt, bias, res, C, M, N, K, Kp, act);
  };
  auto ln = [&](const float* x, const float* g, const float* b, float* y, int rows) {
    ln_kernel<<<rows, 32, 0, stream>>>(x, g, b, y, rows);
  };

  // -------- one-time weight conversions (tiny; ~2.7 MB total) --------
  const _Float16* tok_l = convW(3, F_, DM_);
  const _Float16* tok_i = convW(5, F_, DM_);
  const _Float16 *cw_inp[2], *cw_q[2], *cw_k[2], *cw_v[2], *cw_x[2], *cw_o[2],
                 *cw_f0[2], *cw_f1[2];
  const _Float16 *ew_q[4], *ew_k[4], *ew_v[4], *ew_x[4], *ew_o[4], *ew_f0[4], *ew_f1[4];
  for (int r = 0; r < 2; ++r) {
    int cb = 7 + r * 22;
    cw_inp[r] = convW(cb + 0,  DIN_, DM_);
    cw_q[r]   = convW(cb + 6,  DM_, DM_);
    cw_k[r]   = convW(cb + 8,  DM_, DM_);
    cw_v[r]   = convW(cb + 10, DM_, DM_);
    cw_x[r]   = convW(cb + 12, DM_, H_ * F_);
    cw_o[r]   = convW(cb + 14, DM_, DM_);
    cw_f0[r]  = convW(cb + 18, DM_, DFF_);
    cw_f1[r]  = convW(cb + 20, DFF_, DM_);
    for (int l = 0; l < 2; ++l) {
      int e = r * 2 + l;
      int eb = 51 + e * 18;
      ew_q[e]  = convW(eb + 2,  DM_, DM_);
      ew_k[e]  = convW(eb + 4,  DM_, DM_);
      ew_v[e]  = convW(eb + 6,  DM_, DM_);
      ew_x[e]  = convW(eb + 8,  DM_, H_ * F_);
      ew_o[e]  = convW(eb + 10, DM_, DM_);
      ew_f0[e] = convW(eb + 14, DM_, DFF_);
      ew_f1[e] = convW(eb + 16, DFF_, DM_);
    }
  }
  const _Float16* pre_w = convW(125, DM_, DM_);
  const _Float16* fin_w = convW(127, DM_, 1);

  // 1) xdiff (self + cross), packed f16 rows of 32
  {
    int tot = B_ * L_ * L_;
    xdiff_kernel<<<(tot + 255) / 256, 256, 0, stream>>>(x_lat, x_lat, xdsh, L_, L_);
    tot = B_ * L_ * N_;
    xdiff_kernel<<<(tot + 255) / 256, 256, 0, stream>>>(x_lat, x_inp, xdch, L_, N_);
  }
  // 2) x_token: lat = mean gelu(xds@wxl+b) + mean gelu(xdc@wxi+b)
  token_kernel<<<B_ * L_, 256, 0, stream>>>(xdsh, tok_l, P(4), lat, L_, 0);
  token_kernel<<<B_ * L_, 256, 0, stream>>>(xdch, tok_i, P(6), lat, N_, 1);

  for (int r = 0; r < 2; ++r) {
    int cb = 7 + r * 22;
    // cross-attention block
    gemm(inputs, cw_inp[r], P(cb + 1), nullptr, inp, B_ * N_, DM_, DIN_, 1);
    ln(lat, P(cb + 2), P(cb + 3), latn, B_ * L_);
    ln(inp, P(cb + 4), P(cb + 5), inpn, B_ * N_);
    gemm(latn, cw_q[r], P(cb + 7),  nullptr, qp, B_ * L_, DM_, DM_, 0);
    gemm(inpn, cw_k[r], P(cb + 9),  nullptr, kp, B_ * N_, DM_, DM_, 0);
    gemm(inpn, cw_v[r], P(cb + 11), nullptr, vp, B_ * N_, DM_, DM_, 0);
    gemm(qp,   cw_x[r], P(cb + 13), nullptr, xw, B_ * L_, H_ * F_, DM_, 0);
    attn_kernel<<<B_ * H_ * L_, 256, 0, stream>>>(qp, kp, vp, xw, xdch, ob, L_, N_);
    gemm(ob, cw_o[r], P(cb + 15), lat, lat, B_ * L_, DM_, DM_, 0);
    ln(lat, P(cb + 16), P(cb + 17), latn, B_ * L_);
    gemm(latn, cw_f0[r], P(cb + 19), nullptr, h1, B_ * L_, DFF_, DM_, 1);
    gemm(h1,   cw_f1[r], P(cb + 21), lat, lat, B_ * L_, DM_, DFF_, 0);

    // self-attention encoder layers
    for (int l = 0; l < 2; ++l) {
      int e = r * 2 + l;
      int eb = 51 + e * 18;
      ln(lat, P(eb + 0), P(eb + 1), latn, B_ * L_);
      gemm(latn, ew_q[e], P(eb + 3), nullptr, qp, B_ * L_, DM_, DM_, 0);
      gemm(latn, ew_k[e], P(eb + 5), nullptr, kp, B_ * L_, DM_, DM_, 0);
      gemm(latn, ew_v[e], P(eb + 7), nullptr, vp, B_ * L_, DM_, DM_, 0);
      gemm(qp,   ew_x[e], P(eb + 9), nullptr, xw, B_ * L_, H_ * F_, DM_, 0);
      attn_kernel<<<B_ * H_ * L_, 256, 0, stream>>>(qp, kp, vp, xw, xdsh, ob, L_, L_);
      gemm(ob, ew_o[e], P(eb + 11), lat, lat, B_ * L_, DM_, DM_, 0);
      ln(lat, P(eb + 12), P(eb + 13), latn, B_ * L_);
      gemm(latn, ew_f0[e], P(eb + 15), nullptr, h1, B_ * L_, DFF_, DM_, 1);
      gemm(h1,   ew_f1[e], P(eb + 17), lat, lat, B_ * L_, DM_, DFF_, 0);
    }
  }

  // final head: ln_f -> pre(gelu) -> final
  ln(lat, P(123), P(124), latn, B_ * L_);
  gemm(latn, pre_w, P(126), nullptr, h1, B_ * L_, DM_, DM_, 1);
  gemm(h1, fin_w, P(128), nullptr, (float*)d_out, B_ * L_, 1, DM_, 0);
}
